// TextGenerationModel_3100966387954
// MI455X (gfx1250) — matrix operand order repaired and checked by reading, not compiled, not run on hardware
//
#include <hip/hip_runtime.h>
#include <hip/hip_bf16.h>

typedef __bf16 bf16_t;
typedef __attribute__((ext_vector_type(16))) __bf16 v16bf;
typedef __attribute__((ext_vector_type(8)))  __bf16 bf16x8;
typedef __attribute__((ext_vector_type(8)))  float  v8f;

union V16u { v16bf v; bf16x8 h[2]; };

__device__ __forceinline__ v16bf make_v16(bf16x8 lo, bf16x8 hi) {
  V16u u; u.h[0] = lo; u.h[1] = hi; return u.v;
}

// A fragment, 16x32 bf16 (M x K), row-major A with leading dim ldA.
// Contiguous-k layout used here (matches B): lane<16 holds M=lane, K = k0..k0+15;
//             lane>=16 holds M=lane-16, K = k0+16..k0+31.
__device__ __forceinline__ v16bf load_A_frag(const bf16_t* A, int ldA, int rowBase,
                                             int k0, int lane) {
  int m  = rowBase + (lane & 15);
  int hl = lane >> 4;
  const bf16_t* p = A + (size_t)m * ldA + k0 + hl * 16;
  bf16x8 lo = *(const bf16x8*)p;
  bf16x8 hi = *(const bf16x8*)(p + 8);
  return make_v16(lo, hi);
}

// B fragment, 32x16 bf16 (K x N) where B[k][n] = W[colBase+n][k] (W row-major, ld ldW).
// Contiguous-k layout used here (matches A): lane<16 holds K=k0..k0+15, lane>=16 holds K=k0+16..k0+31, col n=lane&15.
// => one contiguous 32B run of W's row per lane.
__device__ __forceinline__ v16bf load_B_frag(const bf16_t* W, int ldW, int colBase,
                                             int k0, int lane) {
  int n  = colBase + (lane & 15);
  int hl = lane >> 4;
  const bf16_t* p = W + (size_t)n * ldW + k0 + hl * 16;
  bf16x8 lo = *(const bf16x8*)p;
  bf16x8 hi = *(const bf16x8*)(p + 8);
  return make_v16(lo, hi);
}

__device__ __forceinline__ v8f wmma_bf16(v16bf a, v16bf b, v8f c) {
  return __builtin_amdgcn_wmma_f32_16x16x32_bf16(false, a, false, b, (short)0, c,
                                                 false, false);
}

__device__ __forceinline__ float sigmoidf_(float x) {
  return 1.0f / (1.0f + __expf(-x));
}

// ---------------------------------------------------------------- converts
__global__ void __launch_bounds__(256) cvt_f32_bf16(const float* __restrict__ src,
                                                    bf16_t* __restrict__ dst, int n) {
  for (int i = blockIdx.x * 256 + threadIdx.x; i < n; i += gridDim.x * 256)
    dst[i] = (bf16_t)src[i];
}

// gather embedding rows -> bf16 A matrix [4096][256]
__global__ void __launch_bounds__(256) gather_embed(const int* __restrict__ x,
                                                    const float* __restrict__ emb,
                                                    bf16_t* __restrict__ XE) {
  int r = blockIdx.x;            // 4096 rows (t*8+b)
  int v = x[r];
  const float* src = emb + (size_t)v * 256;
  bf16_t* dst = XE + (size_t)r * 256;
  dst[threadIdx.x] = (bf16_t)src[threadIdx.x];   // blockDim = 256
}

// ------------------------------------------------------- pipelined GEMM core
// Each wave: one 16-row M-tile x 4 N-tiles, K stepped in 32-chunks.
// Software double-buffering: loads for chunk kc+1 are issued before the four
// WMMAs of chunk kc, so the matrix ops execute under partial loadcnt waits
// instead of full drains.
template <int KC>
__device__ __forceinline__ void gemm_wave_4n(const bf16_t* __restrict__ A, int ldA,
                                             int mRow, const bf16_t* __restrict__ W,
                                             int ldW, int nColBase, int lane,
                                             v8f acc[4]) {
  v16bf a = load_A_frag(A, ldA, mRow, 0, lane);
  v16bf b0 = load_B_frag(W, ldW, nColBase + 0,  0, lane);
  v16bf b1 = load_B_frag(W, ldW, nColBase + 16, 0, lane);
  v16bf b2 = load_B_frag(W, ldW, nColBase + 32, 0, lane);
  v16bf b3 = load_B_frag(W, ldW, nColBase + 48, 0, lane);
#pragma unroll
  for (int kc = 0; kc < KC; ++kc) {
    v16bf an = a, bn0 = b0, bn1 = b1, bn2 = b2, bn3 = b3;
    if (kc + 1 < KC) {               // compile-time folded (loop fully unrolled)
      int k0 = (kc + 1) * 32;
      an  = load_A_frag(A, ldA, mRow, k0, lane);
      bn0 = load_B_frag(W, ldW, nColBase + 0,  k0, lane);
      bn1 = load_B_frag(W, ldW, nColBase + 16, k0, lane);
      bn2 = load_B_frag(W, ldW, nColBase + 32, k0, lane);
      bn3 = load_B_frag(W, ldW, nColBase + 48, k0, lane);
    }
    acc[0] = wmma_bf16(a, b0, acc[0]);
    acc[1] = wmma_bf16(a, b1, acc[1]);
    acc[2] = wmma_bf16(a, b2, acc[2]);
    acc[3] = wmma_bf16(a, b3, acc[3]);
    a = an; b0 = bn0; b1 = bn1; b2 = bn2; b3 = bn3;
  }
}

// ---------------------------------------------------------------- Zx GEMM
// Zx[4096][2048] = XE[4096][256] @ Wx^T + bcat
__global__ void __launch_bounds__(256) gemm_zx(const bf16_t* __restrict__ XE,
                                               const bf16_t* __restrict__ Wx,
                                               const float* __restrict__ bcat,
                                               float* __restrict__ Zx) {
  int lane = threadIdx.x & 31, wave = threadIdx.x >> 5;
  int job = blockIdx.x * 8 + wave;        // 256 mt x 32 nGroups = 8192 jobs
  int ng = job & 31;
  int mt = job >> 5;
  v8f acc[4] = {};
  gemm_wave_4n<8>(XE, 256, mt * 16, Wx, 256, ng * 64, lane, acc);
  int hl = lane >> 4;
#pragma unroll
  for (int j = 0; j < 4; ++j) {
    int n = ng * 64 + j * 16 + (lane & 15);
    float bias = bcat[n];
#pragma unroll
    for (int r = 0; r < 8; ++r) {
      int m = mt * 16 + r + hl * 8;
      Zx[(size_t)m * 2048 + n] = acc[j][r] + bias;
    }
  }
}

// ---------------------------------------------------------------- recurrence
// 16 persistent WGs; WG wg owns h-cols [wg*32, wg*32+32) across all 4 gates.
// Wh slice (128 z-rows x 512) cached in LDS (128 KB, only possible with CDNA5's
// 320 KB/WGP). Global h state double-buffered in ws; steps chained by
// device-scope arrive counter.
#define REC_NWG 16
__global__ void __launch_bounds__(256) lstm_rec(const bf16_t* __restrict__ Wh,
                                                const float* __restrict__ Zx,
                                                bf16_t* __restrict__ Hg,   // [2][16][512]
                                                bf16_t* __restrict__ HS,   // [4096][512]
                                                unsigned* __restrict__ syncc,
                                                int T) {
  extern __shared__ __align__(16) char smem[];
  bf16_t* shWh = (bf16_t*)smem;                         // [128][512] bf16 = 128 KB
  float*  zbuf = (float*)(smem + 128 * 512 * 2);        // [4][8][32]  = 4 KB
  float*  cbuf = (float*)(smem + 128 * 512 * 2 + 4096); // [8][32]     = 1 KB

  const int tid  = threadIdx.x;
  const int lane = tid & 31;
  const int wave = tid >> 5;          // 8 waves = 8 z-tiles of 16 cols
  const int wg   = blockIdx.x;        // 0..15
  const int hl   = lane >> 4;
  const int nLane = lane & 15;

  // stage Wh slice into LDS: local row lr -> global z-row gate*512 + wg*32 + (lr&31)
  for (int c = tid; c < 128 * 64; c += 256) {   // 64 x uint4 (16B) chunks per row
    int lr = c >> 6, seg = c & 63;
    int zrow = (lr >> 5) * 512 + wg * 32 + (lr & 31);
    ((uint4*)shWh)[lr * 64 + seg] = ((const uint4*)(Wh + (size_t)zrow * 512))[seg];
  }
  cbuf[tid] = 0.0f;    // [8][32] c-state, this WG's slice
  __syncthreads();

  const int gate    = wave >> 1;
  const int colhalf = wave & 1;
  const int tileLr  = wave * 16;                       // local Wh row base of my tile
  const int zcolBase = gate * 512 + wg * 32 + colhalf * 16;

  for (int t = 0; t < T; ++t) {
    const bf16_t* Hread = Hg + (size_t)(t & 1) * 16 * 512;
    v8f acc = {};
#pragma unroll
    for (int kc = 0; kc < 16; ++kc) {
      v16bf a = load_A_frag(Hread, 512, 0, kc * 32, lane);
      v16bf b;  // from LDS, same layout as load_B_frag
      {
        const bf16_t* p = shWh + (size_t)(tileLr + nLane) * 512 + kc * 32 + hl * 16;
        b = make_v16(*(const bf16x8*)p, *(const bf16x8*)(p + 8));
      }
      acc = wmma_bf16(a, b, acc);
    }
    // lanes 0..15 hold the real batch rows m=0..7 in acc[0..7]
    if (hl == 0) {
#pragma unroll
      for (int r = 0; r < 8; ++r) {
        float z = acc[r] + Zx[(size_t)(t * 8 + r) * 2048 + zcolBase + nLane];
        zbuf[gate * 256 + r * 32 + colhalf * 16 + nLane] = z;
      }
    }
    __syncthreads();
    // gate math: 256 threads <-> 8 rows x 32 owned h-cols
    {
      int m = tid >> 5, j = tid & 31;
      float g = tanhf(zbuf[0 * 256 + m * 32 + j]);
      float i = sigmoidf_(zbuf[1 * 256 + m * 32 + j]);
      float f = sigmoidf_(zbuf[2 * 256 + m * 32 + j]);
      float o = sigmoidf_(zbuf[3 * 256 + m * 32 + j]);
      float cc = g * i + cbuf[m * 32 + j] * f;
      cbuf[m * 32 + j] = cc;
      bf16_t hb = (bf16_t)(tanhf(cc) * o);
      Hg[(size_t)((t + 1) & 1) * 16 * 512 + m * 512 + wg * 32 + j] = hb;
      HS[(size_t)(t * 8 + m) * 512 + wg * 32 + j] = hb;
    }
    __threadfence();
    __syncthreads();
    if (tid == 0) {
      atomicAdd(syncc, 1u);
      unsigned target = (unsigned)REC_NWG * (unsigned)(t + 1);
      while (*((volatile unsigned*)syncc) < target)
        __builtin_amdgcn_s_sleep(1);
      __threadfence();
    }
    __syncthreads();
  }
}

// ---------------------------------------------------------------- projection
// out[4096][32000] = HS[4096][512] @ Wph^T + bp
__global__ void __launch_bounds__(256) gemm_logits(const bf16_t* __restrict__ HS,
                                                   const bf16_t* __restrict__ Wph,
                                                   const float* __restrict__ bp,
                                                   float* __restrict__ out) {
  int lane = threadIdx.x & 31, wave = threadIdx.x >> 5;
  int job = blockIdx.x * 8 + wave;     // 256 mt x 500 nGroups = 128000 jobs
  int ng = job % 500;
  int mt = job / 500;
  v8f acc[4] = {};
  gemm_wave_4n<16>(HS, 512, mt * 16, Wph, 512, ng * 64, lane, acc);
  int hl = lane >> 4;
#pragma unroll
  for (int j = 0; j < 4; ++j) {
    int n = ng * 64 + j * 16 + (lane & 15);
    float bias = bp[n];
#pragma unroll
    for (int r = 0; r < 8; ++r) {
      size_t m = (size_t)(mt * 16 + r + hl * 8);
      out[m * 32000 + n] = acc[j][r] + bias;
    }
  }
}

// ---------------------------------------------------------------- softmax
// one WG per row of 32000; online max/sum pass then normalize pass (in place)
__global__ void __launch_bounds__(256) softmax_rows(float* __restrict__ out) {
  __shared__ float rm[256], rs[256];
  float* p = out + (size_t)blockIdx.x * 32000;
  float m = -3.4e38f, s = 0.0f;
  for (int c = threadIdx.x; c < 32000; c += 256) {
    float x = p[c];
    float nm = fmaxf(m, x);
    s = s * __expf(m - nm) + __expf(x - nm);
    m = nm;
  }
  rm[threadIdx.x] = m; rs[threadIdx.x] = s;
  __syncthreads();
  for (int off = 128; off > 0; off >>= 1) {
    if (threadIdx.x < off) {
      float m1 = rm[threadIdx.x], s1 = rs[threadIdx.x];
      float m2 = rm[threadIdx.x + off], s2 = rs[threadIdx.x + off];
      float nm = fmaxf(m1, m2);
      rm[threadIdx.x] = nm;
      rs[threadIdx.x] = s1 * __expf(m1 - nm) + s2 * __expf(m2 - nm);
    }
    __syncthreads();
  }
  float M = rm[0], inv = 1.0f / rs[0];
  for (int c = threadIdx.x; c < 32000; c += 256)
    p[c] = __expf(p[c] - M) * inv;
}

// ---------------------------------------------------------------- launch
extern "C" void kernel_launch(void* const* d_in, const int* in_sizes, int n_in,
                              void* d_out, int out_size, void* d_ws, size_t ws_size,
                              hipStream_t stream) {
  const int*   x   = (const int*)d_in[0];
  const float* emb = (const float*)d_in[1];
  const float* Wgx = (const float*)d_in[2];
  const float* Wgh = (const float*)d_in[3];
  const float* Wix = (const float*)d_in[4];
  const float* Wih = (const float*)d_in[5];
  const float* Wfx = (const float*)d_in[6];
  const float* Wfh = (const float*)d_in[7];
  const float* Wox = (const float*)d_in[8];
  const float* Woh = (const float*)d_in[9];
  const float* bg  = (const float*)d_in[10];
  const float* bi  = (const float*)d_in[11];
  const float* bf_ = (const float*)d_in[12];
  const float* bo  = (const float*)d_in[13];
  const float* Wph = (const float*)d_in[14];
  const float* bp  = (const float*)d_in[15];
  float* out = (float*)d_out;

  char* ws = (char*)d_ws;
  size_t off = 0;
  auto alloc = [&](size_t bytes) {
    char* p = ws + off;
    off = (off + bytes + 255) & ~(size_t)255;
    return p;
  };
  bf16_t*   Wx_b  = (bf16_t*)alloc(2048UL * 256 * 2);
  bf16_t*   Wh_b  = (bf16_t*)alloc(2048UL * 512 * 2);
  bf16_t*   Wph_b = (bf16_t*)alloc(32000UL * 512 * 2);
  bf16_t*   XE    = (bf16_t*)alloc(4096UL * 256 * 2);
  float*    Zx    = (float*)alloc(4096UL * 2048 * 4);
  bf16_t*   HS    = (bf16_t*)alloc(4096UL * 512 * 2);
  float*    bcat  = (float*)alloc(2048UL * 4);
  bf16_t*   Hg    = (bf16_t*)alloc(2UL * 16 * 512 * 2);
  unsigned* syncc = (unsigned*)alloc(256);

  // reset per-call state (graph replays must be self-contained)
  hipMemsetAsync(Hg, 0, 2UL * 16 * 512 * 2, stream);
  hipMemsetAsync(syncc, 0, 256, stream);
  hipMemcpyAsync(bcat + 0,    bg,  512 * 4, hipMemcpyDeviceToDevice, stream);
  hipMemcpyAsync(bcat + 512,  bi,  512 * 4, hipMemcpyDeviceToDevice, stream);
  hipMemcpyAsync(bcat + 1024, bf_, 512 * 4, hipMemcpyDeviceToDevice, stream);
  hipMemcpyAsync(bcat + 1536, bo,  512 * 4, hipMemcpyDeviceToDevice, stream);

  // bf16 weight staging (concat order: g, i, f, o)
  cvt_f32_bf16<<<512, 256, 0, stream>>>(Wgx, Wx_b + 0 * 512 * 256, 512 * 256);
  cvt_f32_bf16<<<512, 256, 0, stream>>>(Wix, Wx_b + 1 * 512 * 256, 512 * 256);
  cvt_f32_bf16<<<512, 256, 0, stream>>>(Wfx, Wx_b + 2 * 512 * 256, 512 * 256);
  cvt_f32_bf16<<<512, 256, 0, stream>>>(Wox, Wx_b + 3 * 512 * 256, 512 * 256);
  cvt_f32_bf16<<<1024, 256, 0, stream>>>(Wgh, Wh_b + 0 * 512 * 512, 512 * 512);
  cvt_f32_bf16<<<1024, 256, 0, stream>>>(Wih, Wh_b + 1 * 512 * 512, 512 * 512);
  cvt_f32_bf16<<<1024, 256, 0, stream>>>(Wfh, Wh_b + 2 * 512 * 512, 512 * 512);
  cvt_f32_bf16<<<1024, 256, 0, stream>>>(Woh, Wh_b + 3 * 512 * 512, 512 * 512);
  cvt_f32_bf16<<<64000, 256, 0, stream>>>(Wph, Wph_b, 32000 * 512);
  gather_embed<<<4096, 256, 0, stream>>>(x, emb, XE);

  // phase 1: input-side gate pre-GEMM (8192 wave-jobs / 8 waves)
  gemm_zx<<<1024, 256, 0, stream>>>(XE, Wx_b, bcat, Zx);

  // phase 2: recurrence, 16 persistent WGs, 133 KB dynamic LDS each
  size_t recLds = 128 * 512 * 2 + 4096 + 1024;
  lstm_rec<<<REC_NWG, 256, recLds, stream>>>(Wh_b, Zx, Hg, HS, syncc, 512);

  // phase 3: vocab projection straight into d_out (128000 wave-jobs / 8 waves),
  // then in-place softmax
  gemm_logits<<<16000, 256, 0, stream>>>(HS, Wph_b, bp, out);
  softmax_rows<<<4096, 256, 0, stream>>>(out);

  (void)in_sizes; (void)n_in; (void)out_size; (void)ws_size;
}
